// VQVAE_56470230008440
// MI455X (gfx1250) — matrix operand order repaired and checked by reading, not compiled, not run on hardware
//
// CDNA5 (gfx1250) VQ-VAE forward. wave32, WMMA bf16 16x16x32, cluster barriers,
// async global->LDS copies for the GRU hidden-state exchange.
//
// Design notes (MI455X):
//  - All GEMMs (convs via im2col, GRU input projections, GRU recurrence) use
//    V_WMMA_F32_16X16X32_BF16: fp32 accumulate, bf16 operands converted while
//    staging into LDS. Fragment lane layouts follow CDNA5 ISA.
//    GEMM staging has a uniform fast path (full 64x32 / 128x32 tiles, 8
//    contiguous elements per thread, packed bf16x2 LDS stores) and a guarded
//    edge path. All four B fragments are gathered before the WMMA quartet so
//    DS loads overlap matrix issue instead of forcing s_wait_dscnt 0 each op.
//  - GRU recurrence: persistent 16-workgroup cluster kernel. WG c owns 64
//    hidden columns of one direction; its 192x512 bf16 Whh slice (192KB) is
//    pinned in LDS (320KB/WGP on CDNA5). The 16KB h state round-trips through
//    L2 each step via GLOBAL_LOAD_ASYNC_TO_LDS_B128 (ASYNCcnt), and steps are
//    separated by the CDNA5 cluster split barrier: wave0 `s_barrier_signal -3`,
//    all waves `s_barrier_wait -3` (NOP when not cluster-launched, per ISA).
//  - Agent-scope fences (__builtin_amdgcn_fence) publish/acquire h via L2.
//
// ASSUMPTIONS (no runtime available):
//  - d_in order: x, then params flattened as a JAX pytree (dict keys sorted),
//    then hiddens (see loader below).
//  - jax.random.permutation(key(123),8) is not reproducible without threefry;
//    a fixed permutation placeholder is used for the shuffle and idx output.

#include <hip/hip_runtime.h>
#include <hip/hip_bf16.h>
#include <math.h>

typedef __attribute__((ext_vector_type(16))) __bf16 bf16x16;
typedef __attribute__((ext_vector_type(2)))  __bf16 bf16x2;
typedef __attribute__((ext_vector_type(8)))  float  f32x8;

#define BATCH 8
#define MFD   512
#define NGRP  4

// ---------------------------------------------------------------- utilities
__global__ void k_zero(float* __restrict__ p, long n) {
  for (long i = blockIdx.x * (long)blockDim.x + threadIdx.x; i < n;
       i += (long)gridDim.x * blockDim.x) p[i] = 0.f;
}
__global__ void k_copy(float* __restrict__ o, const float* __restrict__ a, long n) {
  for (long i = blockIdx.x * (long)blockDim.x + threadIdx.x; i < n;
       i += (long)gridDim.x * blockDim.x) o[i] = a[i];
}
__global__ void k_add2(float* __restrict__ o, const float* __restrict__ a,
                       const float* __restrict__ b, long n) {
  for (long i = blockIdx.x * (long)blockDim.x + threadIdx.x; i < n;
       i += (long)gridDim.x * blockDim.x) o[i] = a[i] + b[i];
}
__global__ void k_add3(float* __restrict__ o, const float* __restrict__ a,
                       const float* __restrict__ b, const float* __restrict__ c, long n) {
  for (long i = blockIdx.x * (long)blockDim.x + threadIdx.x; i < n;
       i += (long)gridDim.x * blockDim.x) o[i] = a[i] + b[i] + c[i];
}
// (B,C,T) -> (B,T,C)
__global__ void k_nct2ntc(const float* __restrict__ in, float* __restrict__ out,
                          int B, int C, int T) {
  long n = (long)B * C * T;
  for (long i = blockIdx.x * (long)blockDim.x + threadIdx.x; i < n;
       i += (long)gridDim.x * blockDim.x) {
    long c = i % C; long t = (i / C) % T; long b = i / ((long)C * T);
    out[i] = in[((long)b * C + c) * T + t];
  }
}
// (B,T,C) -> (B,C,T)
__global__ void k_ntc2nct(const float* __restrict__ in, float* __restrict__ out,
                          int B, int T, int C) {
  long n = (long)B * T * C;
  for (long i = blockIdx.x * (long)blockDim.x + threadIdx.x; i < n;
       i += (long)gridDim.x * blockDim.x) {
    long c = i % C; long t = (i / C) % T; long b = i / ((long)C * T);
    out[((long)b * C + c) * T + t] = in[i];
  }
}
// out[b][2t+e][c] = in[b][t][c]
__global__ void k_upsample2(const float* __restrict__ in, float* __restrict__ out,
                            int B, int T, int C) {
  long n = (long)B * 2 * T * C;
  for (long i = blockIdx.x * (long)blockDim.x + threadIdx.x; i < n;
       i += (long)gridDim.x * blockDim.x) {
    long c = i % C; long tt = (i / C) % (2 * T); long b = i / ((long)C * 2 * T);
    out[i] = in[((long)b * T + (tt >> 1)) * C + c];
  }
}
// out (B,T,2C): [..c]=x ; [..C+c]=x+g
__global__ void k_concat(const float* __restrict__ x, const float* __restrict__ g,
                         float* __restrict__ out, int B, int T, int C) {
  long n = (long)B * T * 2 * C;
  for (long i = blockIdx.x * (long)blockDim.x + threadIdx.x; i < n;
       i += (long)gridDim.x * blockDim.x) {
    long c = i % (2 * C); long t = (i / (2 * C)) % T; long b = i / ((long)2 * C * T);
    long base = ((long)b * T + t) * C;
    out[i] = (c < C) ? x[base + c] : (x[base + c - C] + g[base + c - C]);
  }
}
// in0 = q*sp + sp ; batch shuffle of sp via packed permutation (4 bits each)
__global__ void k_build_in0(const float* __restrict__ q, const float* __restrict__ sp,
                            float* __restrict__ o, int B, int T, int C, unsigned perm) {
  long n = (long)B * T * C;
  for (long i = blockIdx.x * (long)blockDim.x + threadIdx.x; i < n;
       i += (long)gridDim.x * blockDim.x) {
    long c = i % C; long b = i / ((long)T * C);
    int bb = (int)((perm >> (4 * (int)b)) & 15u);
    float s = sp[(long)bb * C + c];
    o[i] = q[i] * s + s;
  }
}
// im2col: u (B,Tin,C) -> o (B,Tout,C*ks), k = c*ks + tau (matches OIH flatten)
__global__ void k_im2col(const float* __restrict__ u, float* __restrict__ o,
                         int B, int Tin, int C, int Tout, int ks, int stride,
                         int pad, int leaky) {
  long K = (long)C * ks;
  long n = (long)B * Tout * K;
  for (long i = blockIdx.x * (long)blockDim.x + threadIdx.x; i < n;
       i += (long)gridDim.x * blockDim.x) {
    long k = i % K; long t = (i / K) % Tout; long b = i / (K * (long)Tout);
    int c = (int)(k / ks), tau = (int)(k % ks);
    int tin = (int)t * stride + tau - pad;
    float v = 0.f;
    if (tin >= 0 && tin < Tin) v = u[((long)b * Tin + tin) * C + c];
    if (leaky) v = v > 0.f ? v : 0.01f * v;
    o[i] = v;
  }
}

// ---------------------------------------------- CDNA5 async / barrier helpers
__device__ __forceinline__ void async_copy16(unsigned lds_byte_off,
                                             const void* gaddr) {
  // GLOBAL_LOAD_ASYNC_TO_LDS_B128: LDS[VDST + off] <= MEM[VADDR], ASYNCcnt.
  asm volatile("global_load_async_to_lds_b128 %0, %1, off"
               :: "v"(lds_byte_off), "v"(gaddr) : "memory");
}
__device__ __forceinline__ void wait_async0() {
  asm volatile("s_wait_asynccnt 0x0" ::: "memory");
}

// ------------------------------------------------------------ WMMA GEMM
// C[b] (M x N, row stride c_rs) = act( A[b] (M x K, row stride a_rs) * W^T + bias )
// W is (N,K) row-major fp32. bf16 operands, fp32 accumulate.
#define TBM 64
#define TBN 128
#define TBK 32
__global__ __launch_bounds__(256) void k_gemm(
    const float* __restrict__ A, long a_bs, int a_rs,
    const float* __restrict__ W, const float* __restrict__ bias,
    float* __restrict__ C, long c_bs, int c_rs,
    int M, int N, int K, int act) {
  __shared__ __bf16 As[TBM][TBK];
  __shared__ __bf16 Ws[TBN][TBK];
  const int bz = blockIdx.z;
  A += (long)bz * a_bs; C += (long)bz * c_bs;
  const int m0 = blockIdx.y * TBM, n0 = blockIdx.x * TBN;
  const int tid = threadIdx.x, lane = tid & 31, wid = tid >> 5;
  const int wm = wid & 3, wn = wid >> 2;  // wave tile: 16 rows x 64 cols
  f32x8 acc[4];
#pragma unroll
  for (int t = 0; t < 4; ++t)
#pragma unroll
    for (int r = 0; r < 8; ++r) acc[t][r] = 0.f;

  const bool fullM = (m0 + TBM <= M);
  const bool fullN = (n0 + TBN <= N);
  const int srow = tid >> 2;           // staging row 0..63
  const int scol = (tid & 3) * 8;      // staging col 0,8,16,24
  const int arow = wm * 16 + (lane & 15);
  const int kh = (lane >> 4) * 8;   // A: lanes 16-31 hold K=8..15 / 24..31
  const int kb = (lane >> 4) * 16;  // B: lanes 16-31 hold K=8..15 / 24..31 (via kh)

  for (int k0 = 0; k0 < K; k0 += TBK) {
    const bool fullK = (k0 + TBK <= K);
    __builtin_prefetch(W + (long)n0 * K + k0 + TBK, 0, 1);  // global_prefetch_b8
    // ---- stage A tile (64 x 32) ----
    if (fullM && fullK) {
      const float* src = A + (long)(m0 + srow) * a_rs + k0 + scol;
      float v[8];
#pragma unroll
      for (int i = 0; i < 8; ++i) v[i] = src[i];
#pragma unroll
      for (int i = 0; i < 4; ++i) {
        bf16x2 p; p[0] = (__bf16)v[2 * i]; p[1] = (__bf16)v[2 * i + 1];
        *(bf16x2*)&As[srow][scol + 2 * i] = p;
      }
    } else {
      for (int i = tid; i < TBM * TBK; i += 256) {
        int r = i >> 5, c = i & 31;
        int gm = m0 + r, gk = k0 + c;
        float v = (gm < M && gk < K) ? A[(long)gm * a_rs + gk] : 0.f;
        As[r][c] = (__bf16)v;
      }
    }
    // ---- stage W tile (128 x 32) ----
    if (fullN && fullK) {
#pragma unroll
      for (int j = 0; j < 2; ++j) {
        const int r = srow + j * 64;
        const float* src = W + (long)(n0 + r) * K + k0 + scol;
        float v[8];
#pragma unroll
        for (int i = 0; i < 8; ++i) v[i] = src[i];
#pragma unroll
        for (int i = 0; i < 4; ++i) {
          bf16x2 p; p[0] = (__bf16)v[2 * i]; p[1] = (__bf16)v[2 * i + 1];
          *(bf16x2*)&Ws[r][scol + 2 * i] = p;
        }
      }
    } else {
      for (int i = tid; i < TBN * TBK; i += 256) {
        int r = i >> 5, c = i & 31;
        int gn = n0 + r, gk = k0 + c;
        float v = (gn < N && gk < K) ? W[(long)gn * K + gk] : 0.f;
        Ws[r][c] = (__bf16)v;
      }
    }
    __syncthreads();
    // gather A fragment and ALL four B fragments first, then issue the WMMA
    // quartet: lets DS loads drain while matrix ops execute.
    bf16x16 af;
#pragma unroll
    for (int i = 0; i < 8; ++i) {
      af[i]     = As[arow][kh + i];
      af[8 + i] = As[arow][16 + kh + i];
    }
    bf16x16 wf[4];
#pragma unroll
    for (int t = 0; t < 4; ++t) {
      const int col = wn * 64 + t * 16 + (lane & 15);
#pragma unroll
      for (int i = 0; i < 8; ++i) { wf[t][i] = Ws[col][kh + i]; wf[t][8 + i] = Ws[col][16 + kh + i]; }
    }
#pragma unroll
    for (int t = 0; t < 4; ++t) {
      acc[t] = __builtin_amdgcn_wmma_f32_16x16x32_bf16(false, af, false, wf[t],
                                                       (short)0, acc[t], false, false);
    }
    __syncthreads();
  }
  // epilogue: f32 C layout — lanes 0-15 rows 0..7, lanes 16-31 rows 8..15
  const int mbase = (lane >> 4) * 8;
  const int ncol = lane & 15;
#pragma unroll
  for (int t = 0; t < 4; ++t) {
    int gn = n0 + wn * 64 + t * 16 + ncol;
    if (gn >= N) continue;
    float bv = bias ? bias[gn] : 0.f;
#pragma unroll
    for (int r = 0; r < 8; ++r) {
      int gm = m0 + wm * 16 + mbase + r;
      if (gm >= M) continue;
      float v = acc[t][r] + bv;
      if (act == 1) v = v > 0.f ? v : 0.01f * v;
      C[(long)gm * c_rs + gn] = v;
    }
  }
}

// ----------------------------------------------------- GroupNorm (+leaky)
__global__ __launch_bounds__(256) void k_groupnorm(
    float* __restrict__ x, const float* __restrict__ gamma,
    const float* __restrict__ beta, int B, int T, int C, int G, int leaky) {
  const int b = blockIdx.x / G, g = blockIdx.x % G;
  const int cg = C / G;
  const long n = (long)T * cg;
  __shared__ float s1[256], s2[256];
  const int tid = threadIdx.x;
  float sm = 0.f, sq = 0.f;
  for (long i = tid; i < n; i += 256) {
    int t = (int)(i / cg), c = g * cg + (int)(i % cg);
    float v = x[((long)b * T + t) * C + c];
    sm += v; sq += v * v;
  }
  s1[tid] = sm; s2[tid] = sq; __syncthreads();
  for (int s = 128; s > 0; s >>= 1) {
    if (tid < s) { s1[tid] += s1[tid + s]; s2[tid] += s2[tid + s]; }
    __syncthreads();
  }
  float mean = s1[0] / (float)n;
  float var = s2[0] / (float)n - mean * mean;
  float inv = rsqrtf(var + 1e-5f);
  for (long i = tid; i < n; i += 256) {
    int t = (int)(i / cg), c = g * cg + (int)(i % cg);
    long idx = ((long)b * T + t) * C + c;
    float v = (x[idx] - mean) * inv * gamma[c] + beta[c];
    if (leaky) v = v > 0.f ? v : 0.01f * v;
    x[idx] = v;
  }
}

// ------------------------------------------------- encoder norm / quantize
__global__ __launch_bounds__(256) void k_enc_norm(const float* __restrict__ y,
                                                  float* __restrict__ xn,
                                                  int T, int C) {
  const int b = blockIdx.x, tid = threadIdx.x;
  __shared__ float mean_s[64], inv_s[64];
  if (tid < C) {
    float s = 0.f;
    for (int t = 0; t < T; ++t) s += y[((long)b * T + t) * C + tid];
    float m = s / (float)T;
    float ss = 0.f;
    for (int t = 0; t < T; ++t) {
      float d = y[((long)b * T + t) * C + tid] - m; ss += d * d;
    }
    mean_s[tid] = m;
    inv_s[tid] = 1.f / (sqrtf(ss) + 1e-4f);
  }
  __syncthreads();
  for (int t = tid; t < T; t += 256) {
    float nrm = 0.f;
    for (int c = 0; c < C; ++c) {
      float v = (y[((long)b * T + t) * C + c] - mean_s[c]) * inv_s[c];
      nrm += v * v;
    }
    float r = rsqrtf(nrm);
    for (int c = 0; c < C; ++c) {
      float v = (y[((long)b * T + t) * C + c] - mean_s[c]) * inv_s[c];
      xn[((long)b * T + t) * C + c] = v * r;
    }
  }
}
__global__ void k_quantize(const float* __restrict__ xn, const float* __restrict__ Wc,
                           float* __restrict__ q, float* __restrict__ diff,
                           int BT, int C, int ncodes, float scale) {
  int i = blockIdx.x * blockDim.x + threadIdx.x;
  if (i >= BT) return;
  const float* v = xn + (long)i * C;
  int best = 0; float bd = 1e30f;
  for (int n = 0; n < ncodes; ++n) {
    const float* w = Wc + (long)n * C;
    float wn = 0.f, dot = 0.f;
    for (int c = 0; c < C; ++c) { wn += w[c] * w[c]; dot += v[c] * w[c]; }
    float d = -2.f * dot * rsqrtf(wn);  // |v|^2 and |e|^2=1 constant per i
    if (d < bd) { bd = d; best = n; }
  }
  const float* w = Wc + (long)best * C;
  float se = 0.f;
  for (int c = 0; c < C; ++c) {
    float qq = w[c];
    q[(long)i * C + c] = qq;
    float dd = qq - v[c]; se += dd * dd;
  }
  atomicAdd(diff, se * scale);
}
__global__ __launch_bounds__(64) void k_sp(const float* __restrict__ y,
                                           const float* __restrict__ q,
                                           float* __restrict__ sp, int T, int C) {
  const int b = blockIdx.x, tid = threadIdx.x;
  __shared__ float spv[64];
  __shared__ float nrm;
  if (tid < C) {
    float s = 0.f;
    for (int t = 0; t < T; ++t)
      s += y[((long)b * T + t) * C + tid] - q[((long)b * T + t) * C + tid];
    spv[tid] = s / (float)T;
  }
  __syncthreads();
  if (tid == 0) {
    float s = 0.f;
    for (int c = 0; c < C; ++c) s += spv[c] * spv[c];
    nrm = 1.f / ((sqrtf(s) + 1e-4f) * 3.f);
  }
  __syncthreads();
  if (tid < C) sp[(long)b * C + tid] = spv[tid] * nrm;
}

// ------------------------------------------------------ GRU cluster kernel
// 16 WGs (ideally one 16-WG cluster): WG c = (dir = c>>3, slot = c&7) owns
// hidden cols [slot*64, slot*64+64) of direction dir. Whh slice (192x512 bf16)
// lives in LDS for the whole recurrence; 16KB h state is exchanged via L2 with
// GLOBAL_LOAD_ASYNC_TO_LDS_B128 each step, fenced by the cluster barrier.
#define GRU_T 384
#define GRU_SMEM (8 * 192 * 4 + 192 * 4 + 16 * 512 * 4 + 192 * 512 * 2)
__global__ __launch_bounds__(GRU_T) void k_gru(
    const float* __restrict__ whh_f, const float* __restrict__ bhh_f,
    const float* __restrict__ whh_b, const float* __restrict__ bhh_b,
    const float* __restrict__ xp_f, const float* __restrict__ xp_b,  // (8,T,1536)
    const float* __restrict__ h0,                                    // (2,8,512)
    float* __restrict__ hglob,                                       // (2,2,8,512)
    float* __restrict__ ys,                                          // (2,8,T,512)
    int T) {
  extern __shared__ char smem[];
  float* g_lds = (float*)smem;                 // [8][192] gates
  float* bhh_l = g_lds + 8 * 192;              // [192]
  float* h_f32 = bhh_l + 192;                  // [16][512] fp32 (rows 8..15 = 0)
  __bf16* w_lds = (__bf16*)(h_f32 + 16 * 512); // [192][512]
  // dynamic LDS starts at offset 0 (no static LDS in this kernel)
  const unsigned h_lds_base = (unsigned)((char*)h_f32 - smem);

  const int tid = threadIdx.x;
  const int lane = tid & 31;
  const int wv = tid >> 5;        // 0..11, one 16x16 N-tile each
  const int cid = blockIdx.x;     // 0..15
  const int dir = cid >> 3;
  const int slot = cid & 7;

  const float* whh = dir ? whh_b : whh_f;
  const float* bhh = dir ? bhh_b : bhh_f;
  const float* xp = dir ? xp_b : xp_f;

  // preload recurrent weight slice (rows: r/z/n blocks for owned cols)
  for (int i = tid * 8; i < 192 * 512; i += GRU_T * 8) {
    int nl = i >> 9, k = i & 511;
    int row = (nl >> 6) * 512 + slot * 64 + (nl & 63);
    const float* src = whh + (long)row * 512 + k;
#pragma unroll
    for (int j = 0; j < 4; ++j) {
      bf16x2 p; p[0] = (__bf16)src[2 * j]; p[1] = (__bf16)src[2 * j + 1];
      *(bf16x2*)&w_lds[i + 2 * j] = p;
    }
  }
  if (tid < 192) bhh_l[tid] = bhh[(tid >> 6) * 512 + slot * 64 + (tid & 63)];
  for (int i = tid; i < 8 * 512; i += GRU_T) h_f32[8 * 512 + i] = 0.f;
  __syncthreads();

  const int arow = lane & 15;
  const int kh = (lane >> 4) * 8;
  const int kb = (lane >> 4) * 16;
  const int bcol = wv * 16 + (lane & 15);

  for (int t = 0; t < T; ++t) {
    // async-stage h of own direction (rows 0..7), fp32, 16B per lane-op
    const float* hsrc = (t == 0) ? (h0 + (long)dir * 8 * 512)
                                 : (hglob + ((long)((t - 1) & 1) * 2 + dir) * 8 * 512);
    for (int i = tid; i < 1024; i += GRU_T)
      async_copy16(h_lds_base + (unsigned)i * 16u, hsrc + (long)i * 4);
    wait_async0();
    __syncthreads();

    // gates = h @ Whh_slice^T : M=16 (8 live), N=192, K=512
    f32x8 acc;
#pragma unroll
    for (int r = 0; r < 8; ++r) acc[r] = 0.f;
#pragma unroll 4
    for (int kc = 0; kc < 512; kc += 32) {
      bf16x16 af, wf;
#pragma unroll
      for (int i = 0; i < 8; ++i) {
        af[i]     = (__bf16)h_f32[arow * 512 + kc + kh + i];
        af[8 + i] = (__bf16)h_f32[arow * 512 + kc + 16 + kh + i];
      }
#pragma unroll
      for (int i = 0; i < 8; ++i) { wf[i] = w_lds[bcol * 512 + kc + kh + i]; wf[8 + i] = w_lds[bcol * 512 + kc + 16 + kh + i]; }
      acc = __builtin_amdgcn_wmma_f32_16x16x32_bf16(false, af, false, wf,
                                                    (short)0, acc, false, false);
    }
    if (lane < 16) {  // only rows 0..7 are live batch rows
#pragma unroll
      for (int r = 0; r < 8; ++r) g_lds[r * 192 + wv * 16 + lane] = acc[r];
    }
    __syncthreads();

    const int tt = dir ? (T - 1 - t) : t;
    for (int i = tid; i < 8 * 64; i += GRU_T) {
      int b = i >> 6, j = i & 63;
      int cg = slot * 64 + j;
      const float* xpr = xp + ((long)b * T + tt) * 1536;
      float gr = g_lds[b * 192 + j] + bhh_l[j];
      float gz = g_lds[b * 192 + 64 + j] + bhh_l[64 + j];
      float gn = g_lds[b * 192 + 128 + j] + bhh_l[128 + j];
      float r = 1.f / (1.f + expf(-(xpr[cg] + gr)));
      float z = 1.f / (1.f + expf(-(xpr[512 + cg] + gz)));
      float nn = tanhf(xpr[1024 + cg] + r * gn);
      float hnew = (1.f - z) * nn + z * h_f32[b * 512 + cg];
      hglob[(((long)(t & 1) * 2 + dir) * 8 + b) * 512 + cg] = hnew;
      ys[(((long)dir * 8 + b) * T + tt) * 512 + cg] = hnew;
    }
    // publish h, then cluster split-barrier (one signal per WG, all waves wait)
    __builtin_amdgcn_fence(__ATOMIC_RELEASE, "agent");
    __syncthreads();
    if (threadIdx.x < 32) asm volatile("s_barrier_signal -3" ::: "memory");
    asm volatile("s_barrier_wait -3" ::: "memory");
    __builtin_amdgcn_fence(__ATOMIC_ACQUIRE, "agent");
    __syncthreads();
  }
}

__global__ void k_tail(float* __restrict__ out, const float* __restrict__ diff,
                       unsigned perm) {
  int i = threadIdx.x;
  if (i == 0) out[0] = *diff;
  if (i >= 1 && i <= 8) out[i] = (float)((perm >> (4 * (i - 1))) & 15u);
}

// =========================================================== host side
struct GruP { const float *bhh, *bih, *whh, *wih; };
struct GbP {
  const float *ci_b, *ci_w, *co_b, *co_w, *gn_b, *gn_g;
  const float *cb, *cw; GruP gb; const float* gbt; GruP gf; const float* gg;
};

static inline int cdiv(long a, long b) { return (int)((a + b - 1) / b); }
static inline void launch_ew(long n, int& blocks) { blocks = cdiv(n, 256); if (blocks > 65535) blocks = 65535; }

static void gemm(hipStream_t st, const float* A, long a_bs, int a_rs,
                 const float* W, const float* bias, float* C, long c_bs,
                 int c_rs, int M, int N, int K, int act) {
  dim3 g(cdiv(N, TBN), cdiv(M, TBM), BATCH);
  k_gemm<<<g, 256, 0, st>>>(A, a_bs, a_rs, W, bias, C, c_bs, c_rs, M, N, K, act);
}
static void im2col(hipStream_t st, const float* u, float* o, int Tin, int C,
                   int Tout, int ks, int stride, int pad, int leaky) {
  long n = (long)BATCH * Tout * C * ks; int bl; launch_ew(n, bl);
  k_im2col<<<bl, 256, 0, st>>>(u, o, BATCH, Tin, C, Tout, ks, stride, pad, leaky);
}

struct Scratch {
  float *col, *y1, *y2, *xn, *q[3], *sp[3];
  float *h1, *xpf, *xpb, *ys, *r, *cbuf, *rc, *hglob, *gbout, *in0, *xa, *xb, *xt, *diff;
};

static void run_gblock(hipStream_t st, const GbP& p, const float* u, int T,
                       int Cin, int Cout, const float* h0, float* out,
                       const Scratch& S) {
  long nTM = (long)BATCH * T * MFD;
  int bl;
  // ci: conv Cin->512 k3 p1, then GN+leaky
  im2col(st, u, S.col, T, Cin, T, 3, 1, 1, 0);
  gemm(st, S.col, (long)T * Cin * 3, Cin * 3, p.ci_w, p.ci_b, S.h1,
       (long)T * MFD, MFD, T, MFD, Cin * 3, 0);
  k_groupnorm<<<BATCH * NGRP, 256, 0, st>>>(S.h1, p.gn_g, p.gn_b, BATCH, T, MFD, NGRP, 1);
  // GRU input projections (bulk WMMA GEMMs)
  gemm(st, S.h1, (long)T * MFD, MFD, p.gf.wih, p.gf.bih, S.xpf,
       (long)T * 1536, 1536, T, 1536, MFD, 0);
  gemm(st, S.h1, (long)T * MFD, MFD, p.gb.wih, p.gb.bih, S.xpb,
       (long)T * 1536, 1536, T, 1536, MFD, 0);
  // recurrence: 16-WG cluster kernel, Whh pinned in LDS
  k_gru<<<16, GRU_T, GRU_SMEM, st>>>(p.gf.whh, p.gf.bhh, p.gb.whh, p.gb.bhh,
                                     S.xpf, S.xpb, h0, S.hglob, S.ys, T);
  // r = yf + yb
  launch_ew(nTM, bl);
  k_add2<<<bl, 256, 0, st>>>(S.r, S.ys, S.ys + (long)BATCH * T * MFD, nTM);
  // grouped conv (G=4) on r, then GN+leaky
  im2col(st, S.r, S.col, T, MFD, T, 3, 1, 1, 0);
  for (int g = 0; g < 4; ++g) {
    gemm(st, S.col + (long)g * 384, (long)T * 1536, 1536,
         p.cw + (long)g * 128 * 384, p.cb + g * 128, S.cbuf + g * 128,
         (long)T * MFD, MFD, T, 128, 384, 0);
  }
  k_groupnorm<<<BATCH * NGRP, 256, 0, st>>>(S.cbuf, p.gg, p.gbt, BATCH, T, MFD, NGRP, 1);
  // rc = h + r + c
  k_add3<<<bl, 256, 0, st>>>(S.rc, S.h1, S.r, S.cbuf, nTM);
  // co: conv 512->Cout k3 p1
  im2col(st, S.rc, S.col, T, MFD, T, 3, 1, 1, 0);
  gemm(st, S.col, (long)T * MFD * 3, MFD * 3, p.co_w, p.co_b, out,
       (long)T * Cout, Cout, T, Cout, MFD * 3, 0);
}

extern "C" void kernel_launch(void* const* d_in, const int* in_sizes, int n_in,
                              void* d_out, int out_size, void* d_ws, size_t ws_size,
                              hipStream_t stream) {
  (void)in_sizes; (void)n_in; (void)out_size; (void)ws_size;
  const float* x = (const float*)d_in[0];
  int pi = 1;
  auto nxt = [&]() { return (const float*)d_in[pi++]; };
  auto loadGru = [&](GruP& g) { g.bhh = nxt(); g.bih = nxt(); g.whh = nxt(); g.wih = nxt(); };
  auto loadGb = [&](GbP& g) {
    g.ci_b = nxt(); g.ci_w = nxt(); g.co_b = nxt(); g.co_w = nxt();
    g.gn_b = nxt(); g.gn_g = nxt();
    g.cb = nxt(); g.cw = nxt(); loadGru(g.gb); g.gbt = nxt(); loadGru(g.gf); g.gg = nxt();
  };
  GbP blocks[3], refine[3], res[3];
  for (int i = 0; i < 3; ++i) loadGb(blocks[i]);
  for (int i = 0; i < 3; ++i) loadGb(refine[i]);
  for (int i = 0; i < 3; ++i) loadGb(res[i]);
  const float* embed[3]; for (int i = 0; i < 3; ++i) embed[i] = nxt();
  struct EncP { const float *c1b, *c1w, *c2b, *c2w; } enc[3];
  for (int i = 0; i < 3; ++i) { enc[i].c1b = nxt(); enc[i].c1w = nxt(); enc[i].c2b = nxt(); enc[i].c2w = nxt(); }
  const float* hiddens = nxt();

  // bump allocator on d_ws (identical layout every call -> deterministic)
  char* base = (char*)d_ws; size_t off = 0;
  auto alloc = [&](long n) { float* p = (float*)(base + off); off += ((size_t)n * 4 + 255) & ~(size_t)255; return p; };
  Scratch S;
  S.col = alloc(6291456); S.y1 = alloc(1048576); S.y2 = alloc(81920); S.xn = alloc(81920);
  S.q[0] = alloc(81920); S.q[1] = alloc(20480); S.q[2] = alloc(5120);
  S.sp[0] = alloc(320); S.sp[1] = alloc(160); S.sp[2] = alloc(80);
  S.h1 = alloc(2097152); S.xpf = alloc(6291456); S.xpb = alloc(6291456);
  S.ys = alloc(4194304); S.r = alloc(2097152); S.cbuf = alloc(2097152);
  S.rc = alloc(2097152); S.hglob = alloc(16384); S.gbout = alloc(327680);
  S.in0 = alloc(81920); S.xa = alloc(327680); S.xb = alloc(327680);
  S.xt = alloc(327680); S.diff = alloc(1);

  int bl;
  k_zero<<<1, 256, 0, stream>>>(S.diff, 1);

  // ---------------- encode ----------------
  launch_ew((long)BATCH * 80 * 512, bl);
  k_nct2ntc<<<bl, 256, 0, stream>>>(x, S.xt, BATCH, 80, 512);
  const float* cur = S.xt;
  int Cin = 80, Tin = 512;
  for (int i = 0; i < 3; ++i) {
    int Tout = Tin / 2, Cout = Cin / 2;
    im2col(stream, cur, S.col, Tin, Cin, Tout, 4, 2, 1, 0);
    gemm(stream, S.col, (long)Tout * Cin * 4, Cin * 4, enc[i].c1w, enc[i].c1b,
         S.y1, (long)Tout * MFD, MFD, Tout, MFD, Cin * 4, 0);
    im2col(stream, S.y1, S.col, Tout, MFD, Tout, 3, 1, 1, /*leaky=*/1);
    gemm(stream, S.col, (long)Tout * MFD * 3, MFD * 3, enc[i].c2w, enc[i].c2b,
         S.y2, (long)Tout * Cout, Cout, Tout, Cout, MFD * 3, 0);
    k_enc_norm<<<BATCH, 256, 0, stream>>>(S.y2, S.xn, Tout, Cout);
    int ncodes = 32 << i;
    float scale = 1.f / (float)((long)BATCH * Tout * Cout);
    k_quantize<<<cdiv((long)BATCH * Tout, 128), 128, 0, stream>>>(
        S.xn, embed[i], S.q[i], S.diff, BATCH * Tout, Cout, ncodes, scale);
    k_sp<<<BATCH, 64, 0, stream>>>(S.y2, S.q[i], S.sp[i], Tout, Cout);
    cur = S.y2; Cin = Cout; Tin = Tout;
  }

  // ---------------- decode x2 ----------------
  unsigned perm_id = 0x76543210u;
  static const int PERM[8] = {2, 6, 1, 7, 0, 5, 3, 4};  // threefry placeholder
  unsigned perm_sh = 0;
  for (int b = 0; b < 8; ++b) perm_sh |= ((unsigned)PERM[b]) << (4 * b);

  for (int pass = 0; pass < 2; ++pass) {
    unsigned perm = pass ? perm_sh : perm_id;
    int hoff = 9 * pass;
    for (int i = 0; i < 3; ++i) {
      int Cq = 10 << i, Tr = 64 << i, se = 2 - i;
      long nr = (long)BATCH * Tr * Cq;
      launch_ew(nr, bl);
      k_build_in0<<<bl, 256, 0, stream>>>(S.q[se], S.sp[se], S.in0, BATCH, Tr, Cq, perm);
      const float* h0 = hiddens + (long)(hoff + 3 * i) * 2 * BATCH * MFD;
      run_gblock(stream, res[i], S.in0, Tr, Cq, Cq, h0, S.gbout, S);
      if (i == 0) k_copy<<<bl, 256, 0, stream>>>(S.xa, S.gbout, nr);
      else        k_add2<<<bl, 256, 0, stream>>>(S.xa, S.xa, S.gbout, nr);
      int Tu = 2 * Tr;
      long nu = (long)BATCH * Tu * Cq;
      launch_ew(nu, bl);
      k_upsample2<<<bl, 256, 0, stream>>>(S.xa, S.xb, BATCH, Tr, Cq);
      const float* h1p = hiddens + (long)(hoff + 3 * i + 1) * 2 * BATCH * MFD;
      run_gblock(stream, blocks[i], S.xb, Tu, Cq, Cq, h1p, S.gbout, S);
      k_add2<<<bl, 256, 0, stream>>>(S.xb, S.xb, S.gbout, nu);
      const float* h2p = hiddens + (long)(hoff + 3 * i + 2) * 2 * BATCH * MFD;
      run_gblock(stream, refine[i], S.xb, Tu, Cq, Cq, h2p, S.gbout, S);
      launch_ew(2 * nu, bl);
      k_concat<<<bl, 256, 0, stream>>>(S.xb, S.gbout, S.xa, BATCH, Tu, Cq);
    }
    // xa is (B,512,80) in (B,T,C); reference returns (B,C,T)
    launch_ew((long)BATCH * 512 * 80, bl);
    k_ntc2nct<<<bl, 256, 0, stream>>>(S.xa, (float*)d_out + (long)pass * BATCH * 80 * 512,
                                      BATCH, 512, 80);
  }
  k_tail<<<1, 32, 0, stream>>>((float*)d_out + 2L * BATCH * 80 * 512, S.diff, perm_sh);
}
